// self_attention_26285199851983
// MI455X (gfx1250) — compile-verified, matrix operand order checked by reading, not run on hardware
//
#include <hip/hip_runtime.h>
#include <cstdint>

// ---------------- problem constants ----------------
constexpr int kB  = 16;
constexpr int kC  = 256;
constexpr int kH  = 48;
constexpr int kW  = 48;
constexpr int kN  = kH * kW;     // 2304
constexpr int kQD = 64;          // == KD
constexpr int kVD = 256;

// ---------------- WMMA types / helpers ----------------
typedef __attribute__((ext_vector_type(16))) __bf16          v16bf;
typedef __attribute__((ext_vector_type(8)))  float           v8f;
typedef __attribute__((ext_vector_type(8)))  unsigned short  u16x8;
typedef __attribute__((ext_vector_type(4)))  float           f32x4;

#define DEV static __device__ __forceinline__

union FragAB {
    v16bf bf;
    u16x8 u[2];
};

DEV unsigned short f2bf(float f) {
    union { float f; unsigned int u; } v; v.f = f;
    unsigned int r = v.u + (0x7FFFu + ((v.u >> 16) & 1u));   // RNE
    return (unsigned short)(r >> 16);
}

DEV v8f vzero() {
    v8f z;
#pragma unroll
    for (int i = 0; i < 8; ++i) z[i] = 0.0f;
    return z;
}

// Load a 16x32 bf16 operand fragment from a K-contiguous matrix:
// per lane two contiguous 16-byte loads, matching the CDNA5 VGPR layout
// (lanes 0-15: K 0..7 & 16..23 ; lanes 16-31: K 8..15 & 24..31).
DEV v16bf load_frag(const unsigned short* __restrict__ base, int ld, int row0, int k0) {
    const int lane = threadIdx.x & 31;
    const int r    = row0 + (lane & 15);
    const int kh   = (lane >> 4) << 3;            // 0 or 8
    const unsigned short* p = base + (size_t)r * ld + k0 + kh;
    FragAB f;
    f.u[0] = *(const u16x8*)(p);
    f.u[1] = *(const u16x8*)(p + 16);
    return f.bf;
}

DEV v8f wmma_bf16(v16bf a, v16bf b, v8f c) {
    return __builtin_amdgcn_wmma_f32_16x16x32_bf16(
        /*neg_a=*/false, a, /*neg_b=*/false, b,
        /*c_mod=*/(short)0, c, /*reuse_a=*/false, /*reuse_b=*/false);
}

// ---------------- kernel 0: f32 -> bf16 convert (weights) ----------------
__global__ void k_f32_to_bf16(const float* __restrict__ src,
                              unsigned short* __restrict__ dst, int n) {
    int i = blockIdx.x * blockDim.x + threadIdx.x;
    if (i < n) dst[i] = f2bf(src[i]);
}

// ---------------- kernel 1: x[B,C,N] f32 -> xT[B,N,C] bf16 ----------------
__global__ void k_transpose_x(const float* __restrict__ x,
                              unsigned short* __restrict__ xT) {
    __shared__ float tile[16][65];                // pad -> no bank conflicts
    const int b  = blockIdx.z;
    const int c0 = blockIdx.y * 16;
    const int n0 = blockIdx.x * 64;
#pragma unroll
    for (int it = 0; it < 4; ++it) {
        int idx = threadIdx.x + it * 256;
        int cl = idx >> 6, nl = idx & 63;
        tile[cl][nl] = x[((size_t)b * kC + c0 + cl) * kN + n0 + nl];
    }
    __syncthreads();
#pragma unroll
    for (int it = 0; it < 4; ++it) {
        int idx = threadIdx.x + it * 256;
        int nl = idx >> 4, cl = idx & 15;
        xT[((size_t)b * kN + n0 + nl) * kC + c0 + cl] = f2bf(tile[cl][nl]);
    }
}

// ---------------- kernel 2: q/k projection -------------------------------
// qT[b,pix,o] = sum_c Wq[o,c]*xT[b,pix,c] + bq[o]; wave0 -> q, wave1 -> k.
__global__ void k_proj_qk(const unsigned short* __restrict__ xT,
                          const unsigned short* __restrict__ Wqb,
                          const unsigned short* __restrict__ Wkb,
                          const float* __restrict__ bq,
                          const float* __restrict__ bk,
                          unsigned short* __restrict__ qT,
                          unsigned short* __restrict__ kT) {
    const int pix0 = blockIdx.x * 16;
    const int b    = blockIdx.y;
    const int wave = threadIdx.x >> 5;
    const int lane = threadIdx.x & 31;
    const unsigned short* Wm   = wave ? Wkb : Wqb;
    const float*          bias = wave ? bk : bq;
    unsigned short*       outp = wave ? kT : qT;
    const unsigned short* xTb = xT + ((size_t)b * kN + pix0) * kC;

    v8f acc[4];
#pragma unroll
    for (int t = 0; t < 4; ++t) acc[t] = vzero();

    for (int ks = 0; ks < kC / 32; ++ks) {
        const int k0 = ks * 32;
        v16bf bf = load_frag(xTb, kC, 0, k0);
#pragma unroll
        for (int t = 0; t < 4; ++t) {
            v16bf af = load_frag(Wm, kC, t * 16, k0);
            acc[t] = wmma_bf16(af, bf, acc[t]);
        }
    }
    const int mb  = (lane >> 4) << 3;
    const int pix = pix0 + (lane & 15);
#pragma unroll
    for (int t = 0; t < 4; ++t) {
        u16x8 o8;
#pragma unroll
        for (int v = 0; v < 8; ++v)
            o8[v] = f2bf(acc[t][v] + bias[t * 16 + mb + v]);
        *(u16x8*)(outp + ((size_t)b * kN + pix) * kQD + t * 16 + mb) = o8;
    }
}

// ---------------- kernel 3: v projection ---------------------------------
// v[b,vd,pix] = sum_c Wv[vd,c]*xT[b,pix,c] + bv[vd]; stores pix-contiguous.
__global__ void k_proj_v(const unsigned short* __restrict__ xT,
                         const unsigned short* __restrict__ Wvb,
                         const float* __restrict__ bv,
                         unsigned short* __restrict__ vbuf) {
    const int pix0 = blockIdx.x * 16;
    const int b    = blockIdx.y;
    const int wave = threadIdx.x >> 5;   // 4 waves, 4 vd-tiles each
    const int lane = threadIdx.x & 31;
    const unsigned short* xTb = xT + ((size_t)b * kN + pix0) * kC;

    v8f acc[4];
#pragma unroll
    for (int t = 0; t < 4; ++t) acc[t] = vzero();

    for (int ks = 0; ks < kC / 32; ++ks) {
        const int k0 = ks * 32;
        v16bf af = load_frag(xTb, kC, 0, k0);
#pragma unroll
        for (int t = 0; t < 4; ++t) {
            v16bf bf = load_frag(Wvb, kC, (wave * 4 + t) * 16, k0);
            acc[t] = wmma_bf16(af, bf, acc[t]);
        }
    }
    const int mb = (lane >> 4) << 3;     // pix offset within tile
#pragma unroll
    for (int t = 0; t < 4; ++t) {
        const int vd = (wave * 4 + t) * 16 + (lane & 15);
        const float bias = bv[vd];
        u16x8 o8;
#pragma unroll
        for (int v = 0; v < 8; ++v) o8[v] = f2bf(acc[t][v] + bias);
        *(u16x8*)(vbuf + ((size_t)b * kVD + vd) * kN + pix0 + mb) = o8;
    }
}

// ---------------- kernel 4: energy E[b,i,j] = <q[:,i], k[:,j]> -----------
// Computed as D[m=j, n=i] so the big f32 store is j-contiguous (32B/lane).
__global__ void k_energy(const unsigned short* __restrict__ qT,
                         const unsigned short* __restrict__ kT,
                         float* __restrict__ E) {
    const int i0   = blockIdx.x * 16;
    const int b    = blockIdx.y;
    const int wave = threadIdx.x >> 5;   // 4 waves split the j range
    const int lane = threadIdx.x & 31;

    const unsigned short* qTb = qT + ((size_t)b * kN + i0) * kQD;
    const unsigned short* kTb = kT + (size_t)b * kN * kQD;
    v16bf bf0 = load_frag(qTb, kQD, 0, 0);
    v16bf bf1 = load_frag(qTb, kQD, 0, 32);

    const int mb = (lane >> 4) << 3;
    const int ii = i0 + (lane & 15);
    float* Erow = E + ((size_t)b * kN + ii) * kN;

    const int jt0 = wave * (kN / 16 / 4);          // 36 tiles per wave
    for (int jt = jt0; jt < jt0 + kN / 16 / 4; ++jt) {
        const int j0 = jt * 16;
        v16bf a0 = load_frag(kTb, kQD, j0, 0);
        v16bf a1 = load_frag(kTb, kQD, j0, 32);
        v8f acc = vzero();
        acc = wmma_bf16(a0, bf0, acc);
        acc = wmma_bf16(a1, bf1, acc);
        f32x4 lo, hi;
#pragma unroll
        for (int v = 0; v < 4; ++v) { lo[v] = acc[v]; hi[v] = acc[4 + v]; }
        *(f32x4*)(Erow + j0 + mb)     = lo;
        *(f32x4*)(Erow + j0 + mb + 4) = hi;
    }
}

// ---------------- kernel 5: row softmax ----------------------------------
__global__ void k_softmax(float* __restrict__ E,
                          unsigned short* __restrict__ attnbf) {
    const size_t row = blockIdx.x;                 // b*N + i
    float* Erow = E + row * kN;
    unsigned short* Arow = attnbf + row * kN;
    __shared__ float red[8];
    const int tid = threadIdx.x;

    float vals[kN / 256];                          // 9 per thread
    float m = -1e30f;
#pragma unroll
    for (int t = 0; t < kN / 256; ++t) {
        vals[t] = Erow[tid + 256 * t];
        m = fmaxf(m, vals[t]);
    }
#pragma unroll
    for (int off = 16; off; off >>= 1) m = fmaxf(m, __shfl_xor(m, off, 32));
    if ((tid & 31) == 0) red[tid >> 5] = m;
    __syncthreads();
    float rowmax = red[0];
#pragma unroll
    for (int i = 1; i < 8; ++i) rowmax = fmaxf(rowmax, red[i]);
    __syncthreads();

    float s = 0.0f;
#pragma unroll
    for (int t = 0; t < kN / 256; ++t) {
        vals[t] = __expf(vals[t] - rowmax);
        s += vals[t];
    }
#pragma unroll
    for (int off = 16; off; off >>= 1) s += __shfl_xor(s, off, 32);
    if ((tid & 31) == 0) red[tid >> 5] = s;
    __syncthreads();
    float rowsum = 0.0f;
#pragma unroll
    for (int i = 0; i < 8; ++i) rowsum += red[i];
    const float inv = 1.0f / rowsum;
#pragma unroll
    for (int t = 0; t < kN / 256; ++t) {
        const float a = vals[t] * inv;
        Erow[tid + 256 * t] = a;
        Arow[tid + 256 * t] = f2bf(a);
    }
}

// ---------------- kernel 6: O^T[b,i,vd] = sum_j attn[b,i,j]*v[b,vd,j] ----
// The FLOP-heavy GEMM (K=2304). The 16-row attn slab is streamed into LDS
// with gfx1250 async DMA (GLOBAL_LOAD_ASYNC_TO_LDS_B128, ASYNCcnt), double
// buffered in 256-wide K-panels so the DMA of panel p+1 overlaps the WMMAs
// of panel p. All 4 waves share the LDS copy (no redundant VMEM reads).
// The output tile bounces through LDS for fully coalesced b128 stores.
constexpr int kKP   = 256;            // K-panel width (elements); 9 panels
constexpr int kLROW = kKP + 8;        // padded LDS row stride -> conflict-free

__global__ void k_attn_v(const unsigned short* __restrict__ attnbf,
                         const unsigned short* __restrict__ vbuf,
                         unsigned short* __restrict__ OT) {
    __shared__ unsigned short sA[2][16 * kLROW];   // 2 x 8448 B
    const int i0   = blockIdx.x * 16;
    const int b    = blockIdx.y;
    const int tid  = threadIdx.x;
    const int wave = tid >> 5;        // 4 waves x 4 vd-tiles
    const int lane = tid & 31;
    const unsigned short* Ab = attnbf + ((size_t)b * kN + i0) * kN;
    const unsigned short* Vb = vbuf + (size_t)b * kVD * kN;

    // Async-stage one 16 x 256 bf16 panel: 512 16B chunks, 4 per thread.
    // Each iteration is ONE vector async instruction per wave (32 x 16B).
    auto stage_panel = [&](int p, int buf) {
#pragma unroll
        for (int it = 0; it < 4; ++it) {
            const int idx = tid + it * 128;
            const int row = idx >> 5;             // 32 chunks per row
            const int ch  = idx & 31;
            const unsigned lds =
                (unsigned)(uintptr_t)(&sA[buf][row * kLROW + ch * 8]);
            const unsigned short* g = Ab + (size_t)row * kN + p * kKP + ch * 8;
            asm volatile("global_load_async_to_lds_b128 %0, %1, off"
                         :: "v"(lds), "v"(g) : "memory");
        }
    };

    v8f acc[4];
#pragma unroll
    for (int t = 0; t < 4; ++t) acc[t] = vzero();

    stage_panel(0, 0);
    asm volatile("s_wait_asynccnt 0x0" ::: "memory");
    __syncthreads();

    for (int p = 0; p < kN / kKP; ++p) {
        const int buf = p & 1;
        if (p + 1 < kN / kKP) stage_panel(p + 1, buf ^ 1);   // overlap DMA
        const unsigned short* Apan = &sA[buf][0];
#pragma unroll
        for (int ks = 0; ks < kKP / 32; ++ks) {
            const int k0l = ks * 32;
            // A fragment from LDS: two ds_load_b128 per lane
            const unsigned short* pa =
                Apan + (lane & 15) * kLROW + k0l + ((lane >> 4) << 3);
            FragAB fa;
            fa.u[0] = *(const u16x8*)(pa);
            fa.u[1] = *(const u16x8*)(pa + 16);
            const int k0 = p * kKP + k0l;
#pragma unroll
            for (int t = 0; t < 4; ++t) {
                v16bf bf = load_frag(Vb, kN, (wave * 4 + t) * 16, k0);
                acc[t] = wmma_bf16(fa.bf, bf, acc[t]);
            }
        }
        // next panel's DMA must be in LDS before anyone reads it
        asm volatile("s_wait_asynccnt 0x0" ::: "memory");
        __syncthreads();
    }

    // ---- epilogue: bounce 16x256 tile through LDS, store coalesced ----
    unsigned short* sO = &sA[0][0];   // reuse staging buffer (post-barrier)
    const int mb = (lane >> 4) << 3;
#pragma unroll
    for (int t = 0; t < 4; ++t) {
        const int vd = (wave * 4 + t) * 16 + (lane & 15);
#pragma unroll
        for (int v = 0; v < 8; ++v)
            sO[(mb + v) * kLROW + vd] = f2bf(acc[t][v]);
    }
    __syncthreads();
#pragma unroll
    for (int it = 0; it < 4; ++it) {
        const int idx = tid + it * 128;
        const int row = idx >> 5;                 // 32 chunks of 16B per row
        const int ch  = idx & 31;
        u16x8 d = *(const u16x8*)(&sO[row * kLROW + ch * 8]);
        *(u16x8*)(OT + ((size_t)b * kN + i0 + row) * kVD + ch * 8) = d;
    }
}

// ---------------- kernel 7: out = gamma*(Wo @ O + bo) + x ----------------
__global__ void k_out(const unsigned short* __restrict__ OT,
                      const unsigned short* __restrict__ Wob,
                      const float* __restrict__ bo,
                      const float* __restrict__ gamma,
                      const float* __restrict__ x,
                      float* __restrict__ out) {
    const int i0   = blockIdx.x * 16;
    const int b    = blockIdx.y;
    const int wave = threadIdx.x >> 5;   // 4 waves x 4 c-tiles
    const int lane = threadIdx.x & 31;
    const unsigned short* Ob = OT + ((size_t)b * kN + i0) * kVD;

    v8f acc[4];
#pragma unroll
    for (int t = 0; t < 4; ++t) acc[t] = vzero();

    for (int ks = 0; ks < kVD / 32; ++ks) {
        const int k0 = ks * 32;
        v16bf af = load_frag(Ob, kVD, 0, k0);
#pragma unroll
        for (int t = 0; t < 4; ++t) {
            v16bf bf = load_frag(Wob, kVD, (wave * 4 + t) * 16, k0);
            acc[t] = wmma_bf16(af, bf, acc[t]);
        }
    }
    const float g  = gamma[0];
    const int   mb = (lane >> 4) << 3;
#pragma unroll
    for (int t = 0; t < 4; ++t) {
        const int c = (wave * 4 + t) * 16 + (lane & 15);
        const size_t base = ((size_t)b * kC + c) * kN + i0 + mb;
        const float bias = bo[c];
        f32x4 x0 = *(const f32x4*)(x + base);
        f32x4 x1 = *(const f32x4*)(x + base + 4);
        f32x4 r0, r1;
#pragma unroll
        for (int v = 0; v < 4; ++v) {
            r0[v] = g * (acc[t][v]     + bias) + x0[v];
            r1[v] = g * (acc[t][4 + v] + bias) + x1[v];
        }
        *(f32x4*)(out + base)     = r0;
        *(f32x4*)(out + base + 4) = r1;
    }
}

// ---------------- host launcher ------------------------------------------
extern "C" void kernel_launch(void* const* d_in, const int* in_sizes, int n_in,
                              void* d_out, int out_size, void* d_ws, size_t ws_size,
                              hipStream_t stream) {
    (void)in_sizes; (void)n_in; (void)out_size; (void)ws_size;
    const float* x     = (const float*)d_in[0];
    const float* Wq    = (const float*)d_in[1];
    const float* bq    = (const float*)d_in[2];
    const float* Wk    = (const float*)d_in[3];
    const float* bk    = (const float*)d_in[4];
    const float* Wv    = (const float*)d_in[5];
    const float* bv    = (const float*)d_in[6];
    const float* Wo    = (const float*)d_in[7];
    const float* bo    = (const float*)d_in[8];
    const float* gamma = (const float*)d_in[9];

    float* out      = (float*)d_out;                         // [B,C,H,W]
    float* attn_out = out + (size_t)kB * kC * kN;            // [B,N,N] f32

    // workspace carve (bytes; all offsets 256B aligned). Total ~236 MB.
    char* w = (char*)d_ws;
    size_t off = 0;
    auto carve = [&](size_t bytes) { char* p = w + off; off += bytes; return p; };
    unsigned short* xT     = (unsigned short*)carve((size_t)kB * kN * kC  * 2); // 18.9 MB
    unsigned short* qT     = (unsigned short*)carve((size_t)kB * kN * kQD * 2); //  4.7 MB
    unsigned short* kT     = (unsigned short*)carve((size_t)kB * kN * kQD * 2); //  4.7 MB
    unsigned short* vbuf   = (unsigned short*)carve((size_t)kB * kVD * kN * 2); // 18.9 MB
    unsigned short* attnbf = (unsigned short*)carve((size_t)kB * kN * kN  * 2); // 169.9 MB
    unsigned short* OTbuf  = (unsigned short*)carve((size_t)kB * kN * kVD * 2); // 18.9 MB
    unsigned short* Wqb    = (unsigned short*)carve((size_t)kQD * kC * 2);
    unsigned short* Wkb    = (unsigned short*)carve((size_t)kQD * kC * 2);
    unsigned short* Wvb    = (unsigned short*)carve((size_t)kVD * kC * 2);
    unsigned short* Wob    = (unsigned short*)carve((size_t)kC * kVD * 2);

    // 0) weights -> bf16
    k_f32_to_bf16<<<dim3((kQD * kC + 255) / 256), 256, 0, stream>>>(Wq, Wqb, kQD * kC);
    k_f32_to_bf16<<<dim3((kQD * kC + 255) / 256), 256, 0, stream>>>(Wk, Wkb, kQD * kC);
    k_f32_to_bf16<<<dim3((kVD * kC + 255) / 256), 256, 0, stream>>>(Wv, Wvb, kVD * kC);
    k_f32_to_bf16<<<dim3((kC * kVD + 255) / 256), 256, 0, stream>>>(Wo, Wob, kC * kVD);

    // 1) x -> xT bf16
    k_transpose_x<<<dim3(kN / 64, kC / 16, kB), 256, 0, stream>>>(x, xT);

    // 2) q,k projections (2 waves: q and k)
    k_proj_qk<<<dim3(kN / 16, kB), 64, 0, stream>>>(xT, Wqb, Wkb, bq, bk, qT, kT);

    // 3) v projection
    k_proj_v<<<dim3(kN / 16, kB), 128, 0, stream>>>(xT, Wvb, bv, vbuf);

    // 4) energy -> f32 attention region of d_out
    k_energy<<<dim3(kN / 16, kB), 128, 0, stream>>>(qT, kT, attn_out);

    // 5) softmax in place (f32) + bf16 copy
    k_softmax<<<dim3(kB * kN), 256, 0, stream>>>(attn_out, attnbf);

    // 6) O^T = attn * v^T  (async-LDS double-buffered, K=2304)
    k_attn_v<<<dim3(kN / 16, kB), 128, 0, stream>>>(attnbf, vbuf, OTbuf);

    // 7) output projection + gamma*(.)+x residual
    k_out<<<dim3(kN / 16, kB), 128, 0, stream>>>(OTbuf, Wob, bo, gamma, x, out);
}
